// AttentionA_73864847556825
// MI455X (gfx1250) — matrix operand order repaired and checked by reading, not compiled, not run on hardware
//
#include <hip/hip_runtime.h>

// ---------------------------------------------------------------------------
// CDNA5 / gfx1250 bidirectional cross-attention pipeline, bf16 WMMA everywhere
// Async global->LDS staging (ASYNCcnt) for all contiguous tile copies.
// ---------------------------------------------------------------------------

typedef __bf16 bf16;
typedef bf16  bf16x8 __attribute__((ext_vector_type(8)));
typedef bf16  v16bf  __attribute__((ext_vector_type(16)));
typedef float v8f    __attribute__((ext_vector_type(8)));
typedef float f32x4  __attribute__((ext_vector_type(4)));

static constexpr int B_ = 4, N_ = 1024, M_ = 1024, D_ = 1024, H_ = 16, HD_ = 64;

// Build a 16-element bf16 fragment from two 16-byte LDS loads.
__device__ __forceinline__ v16bf frag_cat(bf16x8 lo, bf16x8 hi) {
  union { v16bf v; bf16x8 h[2]; } u;
  u.h[0] = lo; u.h[1] = hi;
  return u.v;
}

// LDS byte address of a __shared__ location: flat LDS aperture keeps the
// LDS offset in addr[31:0] (ISA 10.2 aperture mapping).
__device__ __forceinline__ unsigned lds_addr_of(const void* p) {
  return (unsigned)(unsigned long long)p;
}

// One thread stages 32 contiguous bytes global->LDS without touching VGPR data:
// IOFFSET is added to BOTH the LDS and global address (ISA 10.7/15.18.3 op 98).
__device__ __forceinline__ void async_copy_b32bytes(unsigned lds, const void* g) {
  asm volatile(
      "global_load_async_to_lds_b128 %0, %1, off\n\t"
      "global_load_async_to_lds_b128 %0, %1, off offset:16"
      :
      : "v"(lds), "v"(g)
      : "memory");
}

__device__ __forceinline__ void async_wait_all() {
  asm volatile("s_wait_asynccnt 0" ::: "memory");
}

// ---------------------------------------------------------------------------
// LayerNorm + cast-to-bf16 (one 1024-wide row per 256-thread block)
// ---------------------------------------------------------------------------
__global__ void __launch_bounds__(256)
ln_cast_kernel(const float* __restrict__ x, const float* __restrict__ w,
               const float* __restrict__ b, bf16* __restrict__ out)
{
  __shared__ float red[2][8];
  const size_t row = blockIdx.x;
  const int t = threadIdx.x, lane = t & 31, wv = t >> 5;

  f32x4 v = ((const f32x4*)(x + row * D_))[t];
  float s  = v.x + v.y + v.z + v.w;
  float s2 = v.x * v.x + v.y * v.y + v.z * v.z + v.w * v.w;
#pragma unroll
  for (int off = 16; off; off >>= 1) {
    s  += __shfl_xor(s,  off, 32);
    s2 += __shfl_xor(s2, off, 32);
  }
  if (lane == 0) { red[0][wv] = s; red[1][wv] = s2; }
  __syncthreads();
  if (wv == 0) {
    float a = (lane < 8) ? red[0][lane] : 0.f;
    float c = (lane < 8) ? red[1][lane] : 0.f;
#pragma unroll
    for (int off = 4; off; off >>= 1) {
      a += __shfl_xor(a, off, 32);
      c += __shfl_xor(c, off, 32);
    }
    if (lane == 0) { red[0][0] = a; red[1][0] = c; }
  }
  __syncthreads();
  const float mu   = red[0][0] * (1.f / D_);
  const float var  = red[1][0] * (1.f / D_) - mu * mu;
  const float rstd = rsqrtf(var + 1e-5f);

  const int c0 = t * 4;
#pragma unroll
  for (int j = 0; j < 4; ++j) {
    float val = (v[j] - mu) * rstd * w[c0 + j] + b[c0 + j];
    out[row * D_ + c0 + j] = (bf16)val;
  }
}

// ---------------------------------------------------------------------------
// f32 -> bf16 cast (weights)
// ---------------------------------------------------------------------------
__global__ void __launch_bounds__(256)
cast_bf16_kernel(const float* __restrict__ in, bf16* __restrict__ out, int n)
{
  const int i = (blockIdx.x * 256 + threadIdx.x) * 4;
  if (i + 3 < n) {
    f32x4 v = *(const f32x4*)(in + i);
#pragma unroll
    for (int j = 0; j < 4; ++j) out[i + j] = (bf16)v[j];
  }
}

// ---------------------------------------------------------------------------
// Generic NN GEMM: C(MxN) = A(MxK,row) * B(KxN,row); bf16 in, bf16 or f32 out
// Tile 128x128x32, 8 waves (4x2), wave computes 32x64 = 2x4 WMMA frags.
// A tile staged via global_load_async_to_lds_b128 (ASYNCcnt path);
// B tile is a transpose scatter so it stays on the VGPR/ds path.
// ---------------------------------------------------------------------------
template <bool OUT_F32>
__global__ void __launch_bounds__(256)
gemm_nn_kernel(const bf16* __restrict__ A, int lda,
               const bf16* __restrict__ B, int ldb,
               void* __restrict__ Cout, int ldc, int K)
{
  constexpr int BM = 128, BN = 128, BK = 32, LS = 40;  // LS: padded LDS stride
  __shared__ __align__(16) bf16 As[BM * LS];
  __shared__ __align__(16) bf16 Bs[BN * LS];

  const int t = threadIdx.x;
  const int lane = t & 31, wave = t >> 5;
  const int l16 = lane & 15, half = lane >> 4;
  const int wr = wave & 3, wc = wave >> 2;
  const int m0 = blockIdx.y * BM, n0 = blockIdx.x * BN;

  const int arow = t >> 1, akoff = (t & 1) * 16;   // A staging: 16 contig bf16
  const int bk = t & 31, bnc = t >> 5;             // B staging: one k-row chunk

  const unsigned lds_a = lds_addr_of(&As[arow * LS + akoff]);

  v8f acc[2][4] = {};

  for (int k0 = 0; k0 < K; k0 += BK) {
    // ---- stage A tile (128x32 row-major): async global->LDS, no VGPR data ----
    const bf16* ag = A + (size_t)(m0 + arow) * lda + (k0 + akoff);
    async_copy_b32bytes(lds_a, ag);
    // ---- stage B tile (32x128) transposed into Bs[n][k] ----
    const bf16* bg = B + (size_t)(k0 + bk) * ldb + (n0 + bnc * 16);
    bf16x8 b0 = *(const bf16x8*)bg;
    bf16x8 b1 = *(const bf16x8*)(bg + 8);
#pragma unroll
    for (int j = 0; j < 8; ++j) Bs[(bnc * 16 + j) * LS + bk] = b0[j];
#pragma unroll
    for (int j = 0; j < 8; ++j) Bs[(bnc * 16 + 8 + j) * LS + bk] = b1[j];

    if (k0 + BK < K) {  // gfx1250 global_prefetch_b8 for next K-tile
      __builtin_prefetch(ag + BK, 0, 1);
      __builtin_prefetch(bg + (size_t)BK * ldb, 0, 1);
    }
    async_wait_all();
    __syncthreads();

    // ISA 16-bit A 16x32 layout: lane=M%16; j<8 -> K=half*8+j, j>=8 -> K=16+half*8+(j-8)
    v16bf afr[2], bfr[4];
#pragma unroll
    for (int i = 0; i < 2; ++i) {
      const bf16* p = &As[(wr * 32 + i * 16 + l16) * LS];
      afr[i] = frag_cat(*(const bf16x8*)(p + half * 8),
                        *(const bf16x8*)(p + 16 + half * 8));
    }
    // ISA 16-bit B 32x16 layout: lane=N%16; j<8 -> K=half*8+j, j>=8 -> K=16+half*8+(j-8)
#pragma unroll
    for (int j = 0; j < 4; ++j) {
      const bf16* p = &Bs[(wc * 64 + j * 16 + l16) * LS];
      bfr[j] = frag_cat(*(const bf16x8*)(p + half * 8),
                        *(const bf16x8*)(p + 16 + half * 8));
    }
#pragma unroll
    for (int i = 0; i < 2; ++i)
#pragma unroll
      for (int j = 0; j < 4; ++j)
        acc[i][j] = __builtin_amdgcn_wmma_f32_16x16x32_bf16(
            false, afr[i], false, bfr[j], (short)0, acc[i][j], false, false);
    __syncthreads();
  }

  // C layout: VGPR r holds M = r (lanes 0-15) / r+8 (lanes 16-31), N = lane%16
#pragma unroll
  for (int i = 0; i < 2; ++i) {
    const int mrow = m0 + wr * 32 + i * 16 + half * 8;
#pragma unroll
    for (int j = 0; j < 4; ++j) {
      const int col = n0 + wc * 64 + j * 16 + l16;
#pragma unroll
      for (int r = 0; r < 8; ++r) {
        const size_t idx = (size_t)(mrow + r) * ldc + col;
        if constexpr (OUT_F32) ((float*)Cout)[idx] = acc[i][j][r];
        else                   ((bf16*)Cout)[idx]  = (bf16)acc[i][j][r];
      }
    }
  }
}

// ---------------------------------------------------------------------------
// Scores (NT GEMM): S[z][n][m] = sum_k q[z,n,k] * ka[z,m,k], K = 64 per head
// Both tiles are contiguous row copies -> fully async global->LDS staging.
// ---------------------------------------------------------------------------
__global__ void __launch_bounds__(256)
scores_kernel(const bf16* __restrict__ q, const bf16* __restrict__ ka,
              bf16* __restrict__ S)
{
  constexpr int BM = 128, BN = 128, LS = 40;
  __shared__ __align__(16) bf16 As[BM * LS];
  __shared__ __align__(16) bf16 Bs[BN * LS];

  const int z = blockIdx.z, bb = z >> 4, hh = z & 15;
  const bf16* A  = q  + (size_t)bb * N_ * D_ + hh * HD_;
  const bf16* Bt = ka + (size_t)bb * M_ * D_ + hh * HD_;  // rows: m, cols: k
  bf16* C = S + (size_t)z * N_ * M_;

  const int t = threadIdx.x;
  const int lane = t & 31, wave = t >> 5;
  const int l16 = lane & 15, half = lane >> 4;
  const int wr = wave & 3, wc = wave >> 2;
  const int n0 = blockIdx.y * BM;      // score rows (n)
  const int c0 = blockIdx.x * BN;      // score cols (m)
  const int arow = t >> 1, akoff = (t & 1) * 16;

  const unsigned lds_a = lds_addr_of(&As[arow * LS + akoff]);
  const unsigned lds_b = lds_addr_of(&Bs[arow * LS + akoff]);

  v8f acc[2][4] = {};

  for (int k0 = 0; k0 < HD_; k0 += 32) {
    async_copy_b32bytes(lds_a, A  + (size_t)(n0 + arow) * D_ + (k0 + akoff));
    async_copy_b32bytes(lds_b, Bt + (size_t)(c0 + arow) * D_ + (k0 + akoff));
    async_wait_all();
    __syncthreads();

    v16bf afr[2], bfr[4];
#pragma unroll
    for (int i = 0; i < 2; ++i) {
      const bf16* p = &As[(wr * 32 + i * 16 + l16) * LS];
      afr[i] = frag_cat(*(const bf16x8*)(p + half * 8),
                        *(const bf16x8*)(p + 16 + half * 8));
    }
#pragma unroll
    for (int j = 0; j < 4; ++j) {
      const bf16* p = &Bs[(wc * 64 + j * 16 + l16) * LS];
      bfr[j] = frag_cat(*(const bf16x8*)(p + half * 8),
                        *(const bf16x8*)(p + 16 + half * 8));
    }
#pragma unroll
    for (int i = 0; i < 2; ++i)
#pragma unroll
      for (int j = 0; j < 4; ++j)
        acc[i][j] = __builtin_amdgcn_wmma_f32_16x16x32_bf16(
            false, afr[i], false, bfr[j], (short)0, acc[i][j], false, false);
    __syncthreads();
  }

#pragma unroll
  for (int i = 0; i < 2; ++i) {
    const int mrow = n0 + wr * 32 + i * 16 + half * 8;
#pragma unroll
    for (int j = 0; j < 4; ++j) {
      const int col = c0 + wc * 64 + j * 16 + l16;
#pragma unroll
      for (int r = 0; r < 8; ++r)
        C[(size_t)(mrow + r) * M_ + col] = (bf16)acc[i][j][r];
    }
  }
}

// ---------------------------------------------------------------------------
// Softmax statistics: one wave per row / one thread per column
// ---------------------------------------------------------------------------
__global__ void __launch_bounds__(256)
row_stats_kernel(const bf16* __restrict__ S, float* __restrict__ rmax,
                 float* __restrict__ rsum)
{
  const int lane = threadIdx.x & 31, wv = threadIdx.x >> 5;
  const size_t row = (size_t)blockIdx.x * 8 + wv;
  const bf16* p = S + row * M_;
  float mx = -3.0e38f;
  for (int i = lane; i < M_; i += 32) mx = fmaxf(mx, (float)p[i]);
#pragma unroll
  for (int off = 16; off; off >>= 1) mx = fmaxf(mx, __shfl_xor(mx, off, 32));
  float sum = 0.f;
  for (int i = lane; i < M_; i += 32) sum += __expf((float)p[i] - mx);
#pragma unroll
  for (int off = 16; off; off >>= 1) sum += __shfl_xor(sum, off, 32);
  if (lane == 0) { rmax[row] = mx; rsum[row] = sum; }
}

__global__ void __launch_bounds__(256)
col_stats_kernel(const bf16* __restrict__ S, float* __restrict__ cmax,
                 float* __restrict__ csum)
{
  const size_t g = (size_t)blockIdx.x * 256 + threadIdx.x;
  const size_t z = g / M_;
  const int m = (int)(g % M_);
  const bf16* p = S + z * (size_t)N_ * M_ + m;
  float mx = -3.0e38f;
  for (int i = 0; i < N_; ++i) mx = fmaxf(mx, (float)p[(size_t)i * M_]);
  float sum = 0.f;
  for (int i = 0; i < N_; ++i) sum += __expf((float)p[(size_t)i * M_] - mx);
  cmax[g] = mx;
  csum[g] = sum;
}

// ---------------------------------------------------------------------------
// Probability GEMM with softmax fused into A staging.
//   TRANS=false: Out[n,hd] += softmax_row(S)[n,m] * V[m,hd]   (x_upd)
//   TRANS=true : Out[m,hd] += softmax_col(S)[m,n] * V[n,hd]   (xa_upd, A = S^T)
// Tile 128x64x32, 8 waves (4x2), wave computes 32x32 = 2x2 WMMA frags.
// ---------------------------------------------------------------------------
template <bool TRANS>
__global__ void __launch_bounds__(256)
attn_apply_kernel(const bf16* __restrict__ S, const float* __restrict__ stat_max,
                  const float* __restrict__ stat_sum, const bf16* __restrict__ V,
                  bf16* __restrict__ Out)
{
  constexpr int BM = 128, BK = 32, LS = 40;
  constexpr int KDIM = TRANS ? N_ : M_;
  constexpr int RDIM = TRANS ? M_ : N_;
  __shared__ __align__(16) bf16 As[BM * LS];
  __shared__ __align__(16) bf16 Bs[64 * LS];

  const int z = blockIdx.z, bb = z >> 4, hh = z & 15;
  const bf16* Sz   = S + (size_t)z * N_ * M_;
  const float* pmax = stat_max + (size_t)z * RDIM;
  const float* psum = stat_sum + (size_t)z * RDIM;
  const bf16* Bsrc = V + (size_t)bb * KDIM * D_ + hh * HD_;
  bf16* C = Out + (size_t)bb * RDIM * D_ + hh * HD_;
  const int m0 = blockIdx.x * BM;

  const int t = threadIdx.x;
  const int lane = t & 31, wave = t >> 5;
  const int l16 = lane & 15, half = lane >> 4;
  const int wr = wave & 3, wc = wave >> 2;

  v8f acc[2][2] = {};

  for (int k0 = 0; k0 < KDIM; k0 += BK) {
    if constexpr (!TRANS) {
      const int row = t >> 1, koff = (t & 1) * 16;
      const float mx  = pmax[m0 + row];
      const float inv = 1.f / psum[m0 + row];
      const bf16* sg = Sz + (size_t)(m0 + row) * M_ + (k0 + koff);
      bf16x8 s0 = *(const bf16x8*)sg;
      bf16x8 s1 = *(const bf16x8*)(sg + 8);
#pragma unroll
      for (int j = 0; j < 8; ++j)
        As[row * LS + koff + j] = (bf16)(__expf((float)s0[j] - mx) * inv);
#pragma unroll
      for (int j = 0; j < 8; ++j)
        As[row * LS + koff + 8 + j] = (bf16)(__expf((float)s1[j] - mx) * inv);
    } else {
      // A[r][k] = S[k0+k][m0+r]; read along m (contiguous), scatter to LDS rows
#pragma unroll
      for (int it = 0; it < 2; ++it) {
        const int lin = t + it * 256;
        const int k = lin & 31, r0 = (lin >> 5) * 8;
        const bf16* sg = Sz + (size_t)(k0 + k) * M_ + (m0 + r0);
        bf16x8 s0 = *(const bf16x8*)sg;
#pragma unroll
        for (int j = 0; j < 8; ++j) {
          const float mx  = pmax[m0 + r0 + j];
          const float inv = 1.f / psum[m0 + r0 + j];
          As[(r0 + j) * LS + k] = (bf16)(__expf((float)s0[j] - mx) * inv);
        }
      }
    }
    // B tile: 32xHD_(64) rows of V, transposed into Bs[n][k]
    if (t < 128) {
      const int k = t & 31, nc = t >> 5;
      const bf16* bg = Bsrc + (size_t)(k0 + k) * D_ + nc * 16;
      bf16x8 b0 = *(const bf16x8*)bg;
      bf16x8 b1 = *(const bf16x8*)(bg + 8);
#pragma unroll
      for (int j = 0; j < 8; ++j) Bs[(nc * 16 + j) * LS + k] = b0[j];
#pragma unroll
      for (int j = 0; j < 8; ++j) Bs[(nc * 16 + 8 + j) * LS + k] = b1[j];
    }
    __syncthreads();

    v16bf afr[2], bfr[2];
#pragma unroll
    for (int i = 0; i < 2; ++i) {
      const bf16* p = &As[(wr * 32 + i * 16 + l16) * LS];
      afr[i] = frag_cat(*(const bf16x8*)(p + half * 8),
                        *(const bf16x8*)(p + 16 + half * 8));
    }
#pragma unroll
    for (int j = 0; j < 2; ++j) {
      const bf16* p = &Bs[(wc * 32 + j * 16 + l16) * LS];
      bfr[j] = frag_cat(*(const bf16x8*)(p + half * 8),
                        *(const bf16x8*)(p + 16 + half * 8));
    }
#pragma unroll
    for (int i = 0; i < 2; ++i)
#pragma unroll
      for (int j = 0; j < 2; ++j)
        acc[i][j] = __builtin_amdgcn_wmma_f32_16x16x32_bf16(
            false, afr[i], false, bfr[j], (short)0, acc[i][j], false, false);
    __syncthreads();
  }

#pragma unroll
  for (int i = 0; i < 2; ++i) {
    const int mrow = m0 + wr * 32 + i * 16 + half * 8;
#pragma unroll
    for (int j = 0; j < 2; ++j) {
      const int col = wc * 32 + j * 16 + l16;
#pragma unroll
      for (int r = 0; r < 8; ++r)
        C[(size_t)(mrow + r) * D_ + col] = (bf16)acc[i][j][r];
    }
  }
}

// ---------------------------------------------------------------------------
// Host-side orchestration
// ---------------------------------------------------------------------------
extern "C" void kernel_launch(void* const* d_in, const int* in_sizes, int n_in,
                              void* d_out, int out_size, void* d_ws, size_t ws_size,
                              hipStream_t stream)
{
  (void)in_sizes; (void)n_in; (void)out_size; (void)ws_size;

  const float* x    = (const float*)d_in[0];
  const float* xa   = (const float*)d_in[1];
  const float* ln_w = (const float*)d_in[2];
  const float* ln_b = (const float*)d_in[3];
  const float* Wq   = (const float*)d_in[4];
  const float* Wkv  = (const float*)d_in[5];
  const float* Wout = (const float*)d_in[6];

  char* base = (char*)d_ws;
  size_t off = 0;
  auto alloc = [&](size_t bytes) -> void* {
    void* p = base + off;
    off += (bytes + 255) & ~(size_t)255;
    return p;
  };

  bf16* xn   = (bf16*)alloc((size_t)B_ * N_ * D_ * 2);
  bf16* xan  = (bf16*)alloc((size_t)B_ * M_ * D_ * 2);
  bf16* wq   = (bf16*)alloc((size_t)D_ * D_ * 2);
  bf16* wkv  = (bf16*)alloc((size_t)D_ * 2 * D_ * 2);
  bf16* wout = (bf16*)alloc((size_t)D_ * D_ * 2);
  bf16* qb   = (bf16*)alloc((size_t)B_ * N_ * D_ * 2);
  bf16* vb   = (bf16*)alloc((size_t)B_ * N_ * D_ * 2);
  bf16* kab  = (bf16*)alloc((size_t)B_ * M_ * D_ * 2);
  bf16* vab  = (bf16*)alloc((size_t)B_ * M_ * D_ * 2);
  bf16* xu   = (bf16*)alloc((size_t)B_ * N_ * D_ * 2);
  bf16* xau  = (bf16*)alloc((size_t)B_ * M_ * D_ * 2);
  float* rmax = (float*)alloc((size_t)B_ * H_ * N_ * 4);
  float* rsum = (float*)alloc((size_t)B_ * H_ * N_ * 4);
  float* cmax = (float*)alloc((size_t)B_ * H_ * M_ * 4);
  float* csum = (float*)alloc((size_t)B_ * H_ * M_ * 4);
  bf16* Sb   = (bf16*)alloc((size_t)B_ * H_ * N_ * M_ * 2);

  // 1) LayerNorm + bf16 cast
  ln_cast_kernel<<<B_ * N_, 256, 0, stream>>>(x,  ln_w, ln_b, xn);
  ln_cast_kernel<<<B_ * M_, 256, 0, stream>>>(xa, ln_w, ln_b, xan);
  // 2) weight casts
  cast_bf16_kernel<<<(D_ * D_) / 1024,     256, 0, stream>>>(Wq,   wq,   D_ * D_);
  cast_bf16_kernel<<<(D_ * 2 * D_) / 1024, 256, 0, stream>>>(Wkv,  wkv,  D_ * 2 * D_);
  cast_bf16_kernel<<<(D_ * D_) / 1024,     256, 0, stream>>>(Wout, wout, D_ * D_);

  // 3) projections (note: x-side K is dead in the reference; skip it)
  dim3 gproj(D_ / 128, (B_ * N_) / 128);
  gemm_nn_kernel<false><<<gproj, 256, 0, stream>>>(xn,  D_, wq,       D_,     qb,  D_, D_);
  gemm_nn_kernel<false><<<gproj, 256, 0, stream>>>(xn,  D_, wkv + D_, 2 * D_, vb,  D_, D_);
  gemm_nn_kernel<false><<<gproj, 256, 0, stream>>>(xan, D_, wkv,      2 * D_, kab, D_, D_);
  gemm_nn_kernel<false><<<gproj, 256, 0, stream>>>(xan, D_, wkv + D_, 2 * D_, vab, D_, D_);

  // 4) scores per (b,h)
  scores_kernel<<<dim3(M_ / 128, N_ / 128, B_ * H_), 256, 0, stream>>>(qb, kab, Sb);

  // 5) softmax stats (both directions over the same scores)
  row_stats_kernel<<<(B_ * H_ * N_) / 8, 256, 0, stream>>>(Sb, rmax, rsum);
  col_stats_kernel<<<(B_ * H_ * M_) / 256, 256, 0, stream>>>(Sb, cmax, csum);

  // 6) probability GEMMs with fused exp/normalize
  attn_apply_kernel<false><<<dim3(N_ / 128, 1, B_ * H_), 256, 0, stream>>>(Sb, rmax, rsum, vab, xu);
  attn_apply_kernel<true ><<<dim3(M_ / 128, 1, B_ * H_), 256, 0, stream>>>(Sb, cmax, csum, vb,  xau);

  // 7) output projections -> fp32 d_out (tuple order: x_upd, xa_upd)
  float* outx  = (float*)d_out;
  float* outxa = outx + (size_t)B_ * N_ * D_;
  gemm_nn_kernel<true><<<gproj, 256, 0, stream>>>(xu,  D_, wout, D_, outx,  D_, D_);
  gemm_nn_kernel<true><<<gproj, 256, 0, stream>>>(xau, D_, wout, D_, outxa, D_, D_);
}
